// DiffusionModel_8529805050323
// MI455X (gfx1250) — matrix operand order repaired and checked by reading, not compiled, not run on hardware
//
#include <hip/hip_runtime.h>
#include <cstdint>
#include <cstddef>

#define NNODES 20000
#define NEDGES 320000

typedef __attribute__((ext_vector_type(16))) __bf16 v16bf;
typedef __attribute__((ext_vector_type(8)))  float  v8f;

struct __align__(16) U4 { unsigned int x, y, z, w; };
struct __align__(8)  U2 { unsigned int x, y; };
union ABf { v16bf v; U4 q[2]; };

__device__ __forceinline__ unsigned short f2bf(float f) {
    unsigned int u = __float_as_uint(f);
    u += 0x7fffu + ((u >> 16) & 1u);
    return (unsigned short)(u >> 16);
}
__device__ __forceinline__ float siluf(float x) { return x / (1.0f + __expf(-x)); }
__device__ __forceinline__ v8f zero8() {
    v8f r;
#pragma unroll
    for (int i = 0; i < 8; ++i) r[i] = 0.0f;
    return r;
}
__device__ __forceinline__ ABf loadB(const unsigned short* p) {
    ABf r; const U4* q = (const U4*)p; r.q[0] = q[0]; r.q[1] = q[1]; return r;
}
// A-fragment (16x32 bf16, M=edge/node rows) from an LDS tile with row stride strideH halves
__device__ __forceinline__ ABf loadA(const unsigned short* tile, int strideH, int kt, int lane) {
    int m  = lane & 15;
    int ko = kt * 32 + ((lane >> 4) << 4);
    const unsigned short* p = tile + m * strideH + ko;
    ABf r; r.q[0] = *(const U4*)p; r.q[1] = *(const U4*)(p + 8); return r;
}
#define WMMA_BF16(A, B, C) \
    __builtin_amdgcn_wmma_f32_16x16x32_bf16(false, (A), false, (B), (short)0, (C), false, false)

// 4-way multi-buffered B-stream GEMM step: 4 loads in flight, then 4 WMMAs
#define GEMM_STEP4(ACC, WS, KT, NT4, NNT, AV)                                          \
    {                                                                                  \
        ABf b0 = loadB((WS) + ((((KT) * (NNT)) + (NT4) + 0) << 9) + lane * 16);        \
        ABf b1 = loadB((WS) + ((((KT) * (NNT)) + (NT4) + 1) << 9) + lane * 16);        \
        ABf b2 = loadB((WS) + ((((KT) * (NNT)) + (NT4) + 2) << 9) + lane * 16);        \
        ABf b3 = loadB((WS) + ((((KT) * (NNT)) + (NT4) + 3) << 9) + lane * 16);        \
        ACC[(NT4) + 0] = WMMA_BF16(AV, b0.v, ACC[(NT4) + 0]);                          \
        ACC[(NT4) + 1] = WMMA_BF16(AV, b1.v, ACC[(NT4) + 1]);                          \
        ACC[(NT4) + 2] = WMMA_BF16(AV, b2.v, ACC[(NT4) + 2]);                          \
        ACC[(NT4) + 3] = WMMA_BF16(AV, b3.v, ACC[(NT4) + 3]);                          \
    }

// ---- pre-swizzle fp32 W[K][N] (row-major) into WMMA B-fragment layout (bf16) ----
__global__ void swizzleB(const float* __restrict__ W, unsigned short* __restrict__ out,
                         int K, int Ncols) {
    int t = blockIdx.x * blockDim.x + threadIdx.x;
    int frag = t >> 5, lane = t & 31;
    int nnt = Ncols >> 4;
    int totalFrags = (K >> 5) * nnt;
    if (frag >= totalFrags) return;
    int kt = frag / nnt, nt = frag - kt * nnt;
    int n = nt * 16 + (lane & 15);
    int kbase = kt * 32 + ((lane >> 4) << 4);
    unsigned short* d = out + ((size_t)frag << 9) + lane * 16;
#pragma unroll
    for (int j = 0; j < 16; ++j) d[j] = f2bf(W[(size_t)(kbase + j) * Ncols + n]);
}

// =====================  fused per-edge MLP  =====================
__global__ __launch_bounds__(128) void edge_kernel(
    const int* __restrict__ eidx, const float* __restrict__ posc,
    const unsigned short* __restrict__ hb,
    const unsigned short* __restrict__ eW1s, const float* __restrict__ eW1last,
    const float* __restrict__ eb1, const float* __restrict__ eg_, const float* __restrict__ ebt,
    const unsigned short* __restrict__ eW2s, const float* __restrict__ eb2,
    const unsigned short* __restrict__ cW1s, const float* __restrict__ cb1,
    const float* __restrict__ cW2,
    float* __restrict__ agg, float* __restrict__ pdelta)
{
    __shared__ __align__(16) unsigned short xbuf[4][4096]; // per-wave: ef(16x128) then x/msg(16x256)
    __shared__ float sbias[7][256];                        // eb1,eg,ebt,eb2,cb1,cW2,eW1[128]
    __shared__ float sdsq[4][16], sedec[4][16], scw[4][16];
    __shared__ int   srow[4][16];

    const int tid = threadIdx.x;
    for (int i = tid; i < 7 * 256; i += 128) {
        int rsel = i >> 8, cc = i & 255;
        float v;
        switch (rsel) {
            case 0: v = eb1[cc]; break;
            case 1: v = eg_[cc]; break;
            case 2: v = ebt[cc]; break;
            case 3: v = eb2[cc]; break;
            case 4: v = cb1[cc]; break;
            case 5: v = cW2[cc]; break;
            default: v = eW1last[cc]; break;
        }
        sbias[rsel][cc] = v;
    }
    __syncthreads();

    const int wave = tid >> 5, lane = tid & 31;
    const int tile = blockIdx.x * 4 + wave;          // 320000/16 = 20000 tiles, exact
    const int m16 = lane & 15, grp = lane >> 4, col = m16;

    // geometry (both half-waves redundantly)
    const int e  = tile * 16 + m16;
    const int er = eidx[e];
    const int ec = eidx[NEDGES + e];
    float px = posc[er * 3 + 0] - posc[ec * 3 + 0];
    float py = posc[er * 3 + 1] - posc[ec * 3 + 1];
    float pz = posc[er * 3 + 2] - posc[ec * 3 + 2];
    float d2 = px * px + py * py + pz * pz;
    float dist = sqrtf(d2);
    float inv = 1.0f / (dist + 1e-6f);
    float rnx = px * inv, rny = py * inv, rnz = pz * inv;
    float edecv = __expf(-dist * 0.2f);
    if (grp == 0) { sdsq[wave][m16] = d2; sedec[wave][m16] = edecv; srow[wave][m16] = er; }

    // gather [h[row] || h[col]] bf16 into LDS tile: 16 edges x 128 halves
    {
        int m = lane >> 1;
        int node = __shfl((lane & 1) ? ec : er, m, 32);
        const U4* src = (const U4*)(hb + (size_t)node * 64);
        U4* dst = (U4*)&xbuf[wave][m * 128 + (lane & 1) * 64];
        U4 tmp[8];
#pragma unroll
        for (int q = 0; q < 8; ++q) tmp[q] = src[q];   // claused loads, one wait
#pragma unroll
        for (int q = 0; q < 8; ++q) dst[q] = tmp[q];
    }
    asm volatile("s_wait_dscnt 0" ::: "memory");

    // ---- GEMM1: x = ef @ eW1   (K=128 via WMMA; bias + rank-1 dist_sq in epilogue) ----
    v8f acc[16];
#pragma unroll
    for (int nt = 0; nt < 16; ++nt) acc[nt] = zero8();
    for (int kt = 0; kt < 4; ++kt) {
        ABf a = loadA(&xbuf[wave][0], 128, kt, lane);
#pragma unroll
        for (int nt4 = 0; nt4 < 16; nt4 += 4) GEMM_STEP4(acc, eW1s, kt, nt4, 16, a.v);
    }
    float dq[8], ed[8]; int rw[8];
#pragma unroll
    for (int r8 = 0; r8 < 8; ++r8) {
        dq[r8] = sdsq[wave][grp * 8 + r8];
        ed[r8] = sedec[wave][grp * 8 + r8];
        rw[r8] = srow[wave][grp * 8 + r8];
    }
    float s1[8], s2[8];
#pragma unroll
    for (int r8 = 0; r8 < 8; ++r8) { s1[r8] = 0.f; s2[r8] = 0.f; }
#pragma unroll
    for (int nt = 0; nt < 16; ++nt) {
        int f = nt * 16 + col;
        float w = sbias[6][f];
        float b1v = sbias[0][f];
#pragma unroll
        for (int r8 = 0; r8 < 8; ++r8) {
            float v = acc[nt][r8] + b1v + dq[r8] * w;
            acc[nt][r8] = v; s1[r8] += v; s2[r8] += v * v;
        }
    }
#pragma unroll
    for (int r8 = 0; r8 < 8; ++r8) {
#pragma unroll
        for (int off = 1; off < 16; off <<= 1) {
            s1[r8] += __shfl_xor(s1[r8], off, 32);
            s2[r8] += __shfl_xor(s2[r8], off, 32);
        }
    }
    float mu[8], rsv[8];
#pragma unroll
    for (int r8 = 0; r8 < 8; ++r8) {
        float m = s1[r8] * (1.0f / 256.0f);
        float var = s2[r8] * (1.0f / 256.0f) - m * m;
        mu[r8] = m; rsv[r8] = rsqrtf(var + 1e-5f);
    }
    // LN -> SiLU -> bf16 into xbuf (stride 256); ef region dead, reads all complete
#pragma unroll
    for (int nt = 0; nt < 16; ++nt) {
        int f = nt * 16 + col;
        float g = sbias[1][f], bb = sbias[2][f];
#pragma unroll
        for (int r8 = 0; r8 < 8; ++r8) {
            float v = (acc[nt][r8] - mu[r8]) * rsv[r8] * g + bb;
            xbuf[wave][(grp * 8 + r8) * 256 + f] = f2bf(siluf(v));
        }
    }

    // ---- GEMM2: msg = silu(x @ eW2 + eb2) * exp(-d/5) ----
#pragma unroll
    for (int nt = 0; nt < 16; ++nt) acc[nt] = zero8();
    for (int kt = 0; kt < 8; ++kt) {
        ABf a = loadA(&xbuf[wave][0], 256, kt, lane);
#pragma unroll
        for (int nt4 = 0; nt4 < 16; nt4 += 4) GEMM_STEP4(acc, eW2s, kt, nt4, 16, a.v);
    }
#pragma unroll
    for (int nt = 0; nt < 16; ++nt) {
        int f = nt * 16 + col;
        float b2v = sbias[3][f];
#pragma unroll
        for (int r8 = 0; r8 < 8; ++r8) {
            float v = siluf(acc[nt][r8] + b2v) * ed[r8];
            xbuf[wave][(grp * 8 + r8) * 256 + f] = f2bf(v);      // msg for GEMM3
            atomicAdd(&agg[(size_t)rw[r8] * 256 + f], v);        // segment_sum(msg)
        }
    }

    // ---- GEMM3: cw = tanh(silu(msg @ cW1 + cb1) @ cW2) ----
#pragma unroll
    for (int nt = 0; nt < 16; ++nt) acc[nt] = zero8();
    for (int kt = 0; kt < 8; ++kt) {
        ABf a = loadA(&xbuf[wave][0], 256, kt, lane);
#pragma unroll
        for (int nt4 = 0; nt4 < 16; nt4 += 4) GEMM_STEP4(acc, cW1s, kt, nt4, 16, a.v);
    }
    float cacc[8];
#pragma unroll
    for (int r8 = 0; r8 < 8; ++r8) cacc[r8] = 0.f;
#pragma unroll
    for (int nt = 0; nt < 16; ++nt) {
        int f = nt * 16 + col;
        float cb = sbias[4][f];
        float w2 = sbias[5][f];
#pragma unroll
        for (int r8 = 0; r8 < 8; ++r8) cacc[r8] += siluf(acc[nt][r8] + cb) * w2;
    }
#pragma unroll
    for (int r8 = 0; r8 < 8; ++r8) {
#pragma unroll
        for (int off = 1; off < 16; off <<= 1) cacc[r8] += __shfl_xor(cacc[r8], off, 32);
    }
    if (col == 0) {
#pragma unroll
        for (int r8 = 0; r8 < 8; ++r8) scw[wave][grp * 8 + r8] = tanhf(cacc[r8]);
    }
    asm volatile("s_wait_dscnt 0" ::: "memory");
    if (grp == 0) {
        float cw = scw[wave][m16];
        atomicAdd(&pdelta[er * 3 + 0], rnx * cw);
        atomicAdd(&pdelta[er * 3 + 1], rny * cw);
        atomicAdd(&pdelta[er * 3 + 2], rnz * cw);
    }
}

// =====================  fused node update  =====================
__global__ __launch_bounds__(128) void node_kernel(
    float* __restrict__ h, const unsigned short* __restrict__ hb,
    const float* __restrict__ agg,
    const unsigned short* __restrict__ nW1s, const float* __restrict__ nb1,
    const unsigned short* __restrict__ nW2s, const float* __restrict__ nb2,
    const float* __restrict__ ng, const float* __restrict__ nbp)
{
    __shared__ __align__(16) unsigned short nbuf[4][16 * 320];
    __shared__ float sb1[256];
    __shared__ float sng[64], snb[64], sb2[64];

    const int tid = threadIdx.x;
    for (int i = tid; i < 256; i += 128) sb1[i] = nb1[i];
    if (tid < 64) { sng[tid] = ng[tid]; snb[tid] = nbp[tid]; sb2[tid] = nb2[tid]; }
    __syncthreads();

    const int wave = tid >> 5, lane = tid & 31;
    const int tile = blockIdx.x * 4 + wave;
    if (tile * 16 >= NNODES) return;                 // wave-uniform skip
    const int grp = lane >> 4, col = lane & 15;

    // stage [hb || bf16(agg)]: 16 nodes x 320 halves
    {
        int m = lane >> 1;
        int node = tile * 16 + m; if (node >= NNODES) node = NNODES - 1;
        const U4* src = (const U4*)(hb + (size_t)node * 64 + (lane & 1) * 32);
        U4* dst = (U4*)&nbuf[wave][m * 320 + (lane & 1) * 32];
        U4 tmp[4];
#pragma unroll
        for (int q = 0; q < 4; ++q) tmp[q] = src[q];
#pragma unroll
        for (int q = 0; q < 4; ++q) dst[q] = tmp[q];
        const float* s = agg + (size_t)node * 256 + (lane & 1) * 128;
        U2* d = (U2*)&nbuf[wave][m * 320 + 64 + (lane & 1) * 128];
#pragma unroll
        for (int c = 0; c < 4; ++c) {
            float4 tv[8];
#pragma unroll
            for (int q = 0; q < 8; ++q) tv[q] = ((const float4*)s)[c * 8 + q];
#pragma unroll
            for (int q = 0; q < 8; ++q) {
                U2 pk;
                pk.x = (unsigned)f2bf(tv[q].x) | ((unsigned)f2bf(tv[q].y) << 16);
                pk.y = (unsigned)f2bf(tv[q].z) | ((unsigned)f2bf(tv[q].w) << 16);
                d[c * 8 + q] = pk;
            }
        }
    }
    asm volatile("s_wait_dscnt 0" ::: "memory");

    // GEMM1n: 320 -> 256
    v8f acc[16];
#pragma unroll
    for (int nt = 0; nt < 16; ++nt) acc[nt] = zero8();
    for (int kt = 0; kt < 10; ++kt) {
        ABf a = loadA(&nbuf[wave][0], 320, kt, lane);
#pragma unroll
        for (int nt4 = 0; nt4 < 16; nt4 += 4) GEMM_STEP4(acc, nW1s, kt, nt4, 16, a.v);
    }
#pragma unroll
    for (int nt = 0; nt < 16; ++nt) {
        int f = nt * 16 + col;
        float b1v = sb1[f];
#pragma unroll
        for (int r8 = 0; r8 < 8; ++r8)
            nbuf[wave][(grp * 8 + r8) * 256 + f] = f2bf(siluf(acc[nt][r8] + b1v));
    }

    // GEMM2n: 256 -> 64
    v8f a2[4];
#pragma unroll
    for (int nt = 0; nt < 4; ++nt) a2[nt] = zero8();
    for (int kt = 0; kt < 8; ++kt) {
        ABf a = loadA(&nbuf[wave][0], 256, kt, lane);
        GEMM_STEP4(a2, nW2s, kt, 0, 4, a.v);
    }

    // residual + LayerNorm(64)
    float vv[4][8], s1[8], s2[8];
#pragma unroll
    for (int r8 = 0; r8 < 8; ++r8) { s1[r8] = 0.f; s2[r8] = 0.f; }
#pragma unroll
    for (int nt = 0; nt < 4; ++nt) {
        int f = nt * 16 + col;
        float b2v = sb2[f];
#pragma unroll
        for (int r8 = 0; r8 < 8; ++r8) {
            int node = tile * 16 + grp * 8 + r8;     // 20000 % 16 == 0: always valid
            float v = h[(size_t)node * 64 + f] + a2[nt][r8] + b2v;
            vv[nt][r8] = v; s1[r8] += v; s2[r8] += v * v;
        }
    }
#pragma unroll
    for (int r8 = 0; r8 < 8; ++r8) {
#pragma unroll
        for (int off = 1; off < 16; off <<= 1) {
            s1[r8] += __shfl_xor(s1[r8], off, 32);
            s2[r8] += __shfl_xor(s2[r8], off, 32);
        }
    }
    float mu[8], rsv[8];
#pragma unroll
    for (int r8 = 0; r8 < 8; ++r8) {
        float m = s1[r8] * (1.0f / 64.0f);
        float var = s2[r8] * (1.0f / 64.0f) - m * m;
        mu[r8] = m; rsv[r8] = rsqrtf(var + 1e-5f);
    }
#pragma unroll
    for (int nt = 0; nt < 4; ++nt) {
        int f = nt * 16 + col;
#pragma unroll
        for (int r8 = 0; r8 < 8; ++r8) {
            int node = tile * 16 + grp * 8 + r8;
            h[(size_t)node * 64 + f] = (vv[nt][r8] - mu[r8]) * rsv[r8] * sng[f] + snb[f];
        }
    }
}

// =====================  small kernels  =====================
__global__ void temb_kernel(const float* t, const float* w1, const float* b1,
                            const float* w2, const float* b2, float* stemb) {
    __shared__ float sa[64];
    int j = threadIdx.x;
    sa[j] = siluf(t[0] * w1[j] + b1[j]);
    __syncthreads();
    float v = b2[j];
    for (int i = 0; i < 64; ++i) v += sa[i] * w2[i * 64 + j];
    stemb[j] = siluf(v);                         // silu(t_emb), reused every layer
}
__global__ void tvec_kernel(const float* stemb, const float* tW, const float* tb, float* tvec) {
    int j = threadIdx.x;
    float v = tb[j];
    for (int i = 0; i < 64; ++i) v += stemb[i] * tW[i * 64 + j];
    tvec[j] = v;
}
__global__ void init_h_kernel(const int* z, const float* emb, float* h) {
    int idx = blockIdx.x * blockDim.x + threadIdx.x;
    if (idx >= NNODES * 64) return;
    h[idx] = emb[z[idx >> 6] * 64 + (idx & 63)];
}
__global__ void h_update_kernel(float* h, unsigned short* hb, const float* tvec) {
    int idx = blockIdx.x * blockDim.x + threadIdx.x;
    if (idx >= NNODES * 64) return;
    float v = h[idx] + tvec[idx & 63];
    h[idx] = v; hb[idx] = f2bf(v);
}
__global__ void pos_update_kernel(float* posc, const float* pd) {
    int idx = blockIdx.x * blockDim.x + threadIdx.x;
    if (idx < NNODES * 3) posc[idx] += pd[idx];
}
__global__ void noise_kernel(const float* posc, const float* pos0, float* out) {
    int idx = blockIdx.x * blockDim.x + threadIdx.x;
    if (idx < NNODES * 3) out[idx] = posc[idx] - pos0[idx];
}
__global__ void hg_reduce_kernel(const float* h, float* hg) {
    __shared__ float sacc[64];
    int tid = threadIdx.x;
    if (tid < 64) sacc[tid] = 0.f;
    __syncthreads();
    int f = tid & 63, sub = tid >> 6;
    int base = blockIdx.x * 256;
    float p = 0.f;
    for (int k = 0; k < 64; ++k) {
        int n = base + sub + k * 4;
        if (n < NNODES) p += h[(size_t)n * 64 + f];
    }
    atomicAdd(&sacc[f], p);
    __syncthreads();
    if (tid < 64) atomicAdd(&hg[tid], sacc[tid]);
}
__global__ void heads_kernel(const float* hg,
    const float* hW1, const float* hb1, const float* hW2, const float* hb2,
    const float* nW1, const float* nb1, const float* nW2, const float* nb2,
    const float* sW1, const float* sb1_, const float* sW2, const float* sb2_,
    float* out) {
    __shared__ float u[64], red[64];
    int f = threadIdx.x;
    u[f] = hg[f] * (1.0f / NNODES);
    __syncthreads();
    auto head = [&](const float* W1, const float* b1, const float* W2, const float* b2) -> float {
        float a = b1[f];
        for (int i = 0; i < 64; ++i) a += u[i] * W1[i * 64 + f];
        a = siluf(a);
        red[f] = a * W2[f];
        __syncthreads();
        for (int off = 32; off > 0; off >>= 1) {
            if (f < off) red[f] += red[f + off];
            __syncthreads();
        }
        float r = red[0] + b2[0];
        __syncthreads();
        return r;
    };
    float vher = head(hW1, hb1, hW2, hb2);
    float ven  = head(nW1, nb1, nW2, nb2);
    float vsy  = head(sW1, sb1_, sW2, sb2_);
    if (f == 0) { out[0] = vher; out[1] = ven; out[2] = 1.0f / (1.0f + __expf(-vsy)); }
}

// =====================  host orchestration  =====================
extern "C" void kernel_launch(void* const* d_in, const int* in_sizes, int n_in,
                              void* d_out, int out_size, void* d_ws, size_t ws_size,
                              hipStream_t stream)
{
    (void)in_sizes; (void)n_in; (void)out_size; (void)ws_size;
    // dict order: z, pos, edge_index, t, emb, tmW1, tmb1, tmW2, tmb2,
    //             6 x [tW,tb,eW1,eb1,eg,ebt,eW2,eb2,nW1,nb1,nW2,nb2,cW1,cb1,cW2,ng,nb],
    //             her{W1,b1,W2,b2}, energy{...}, synth{...}
    const int*   z    = (const int*)d_in[0];
    const float* pos0 = (const float*)d_in[1];
    const int*   eidx = (const int*)d_in[2];
    const float* t    = (const float*)d_in[3];
    const float* emb  = (const float*)d_in[4];
    const float* tmW1 = (const float*)d_in[5];
    const float* tmb1 = (const float*)d_in[6];
    const float* tmW2 = (const float*)d_in[7];
    const float* tmb2 = (const float*)d_in[8];
    auto LP = [&](int l, int k) -> const float* { return (const float*)d_in[9 + l * 17 + k]; };
    const int HER = 9 + 6 * 17;  // 111

    char* base = (char*)d_ws;
    size_t off = 0;
    auto alloc = [&](size_t bytes) -> char* {
        char* p = base + off;
        off += (bytes + 255) & ~(size_t)255;
        return p;
    };
    float* h            = (float*)alloc((size_t)NNODES * 64 * 4);
    unsigned short* hb  = (unsigned short*)alloc((size_t)NNODES * 64 * 2);
    float* agg          = (float*)alloc((size_t)NNODES * 256 * 4);
    float* pdelta       = (float*)alloc((size_t)NNODES * 3 * 4);
    float* posc         = (float*)alloc((size_t)NNODES * 3 * 4);
    float* tvec         = (float*)alloc(64 * 4);
    float* stemb        = (float*)alloc(64 * 4);
    float* hg           = (float*)alloc(64 * 4);
    unsigned short *eW1sw[6], *eW2sw[6], *cW1sw[6], *nW1sw[6], *nW2sw[6];
    for (int l = 0; l < 6; ++l) {
        eW1sw[l] = (unsigned short*)alloc(64  * 1024);  // (128/32)*(256/16) frags
        eW2sw[l] = (unsigned short*)alloc(128 * 1024);
        cW1sw[l] = (unsigned short*)alloc(128 * 1024);
        nW1sw[l] = (unsigned short*)alloc(160 * 1024);
        nW2sw[l] = (unsigned short*)alloc(32  * 1024);
    }

    auto swz = [&](const float* W, unsigned short* outp, int K, int Nc) {
        int thr = (K / 32) * (Nc / 16) * 32;
        swizzleB<<<(thr + 127) / 128, 128, 0, stream>>>(W, outp, K, Nc);
    };
    for (int l = 0; l < 6; ++l) {
        swz(LP(l, 2),  eW1sw[l], 128, 256);   // first 128 rows of eW1
        swz(LP(l, 6),  eW2sw[l], 256, 256);
        swz(LP(l, 12), cW1sw[l], 256, 256);
        swz(LP(l, 8),  nW1sw[l], 320, 256);
        swz(LP(l, 10), nW2sw[l], 256, 64);
    }

    temb_kernel<<<1, 64, 0, stream>>>(t, tmW1, tmb1, tmW2, tmb2, stemb);
    init_h_kernel<<<(NNODES * 64 + 255) / 256, 256, 0, stream>>>(z, emb, h);
    hipMemcpyAsync(posc, pos0, (size_t)NNODES * 3 * 4, hipMemcpyDeviceToDevice, stream);

    for (int l = 0; l < 6; ++l) {
        tvec_kernel<<<1, 64, 0, stream>>>(stemb, LP(l, 0), LP(l, 1), tvec);
        h_update_kernel<<<(NNODES * 64 + 255) / 256, 256, 0, stream>>>(h, hb, tvec);
        hipMemsetAsync(agg, 0, (size_t)NNODES * 256 * 4, stream);
        hipMemsetAsync(pdelta, 0, (size_t)NNODES * 3 * 4, stream);
        edge_kernel<<<NEDGES / 64, 128, 0, stream>>>(
            eidx, posc, hb,
            eW1sw[l], LP(l, 2) + 128 * 256, LP(l, 3), LP(l, 4), LP(l, 5),
            eW2sw[l], LP(l, 7), cW1sw[l], LP(l, 13), LP(l, 14), agg, pdelta);
        pos_update_kernel<<<(NNODES * 3 + 255) / 256, 256, 0, stream>>>(posc, pdelta);
        node_kernel<<<(NNODES / 16 + 3) / 4, 128, 0, stream>>>(
            h, hb, agg, nW1sw[l], LP(l, 9), nW2sw[l], LP(l, 11), LP(l, 15), LP(l, 16));
    }

    noise_kernel<<<(NNODES * 3 + 255) / 256, 256, 0, stream>>>(posc, pos0, (float*)d_out);
    hipMemsetAsync(hg, 0, 64 * 4, stream);
    hg_reduce_kernel<<<(NNODES + 255) / 256, 256, 0, stream>>>(h, hg);
    heads_kernel<<<1, 64, 0, stream>>>(hg,
        (const float*)d_in[HER + 0], (const float*)d_in[HER + 1],
        (const float*)d_in[HER + 2], (const float*)d_in[HER + 3],
        (const float*)d_in[HER + 4], (const float*)d_in[HER + 5],
        (const float*)d_in[HER + 6], (const float*)d_in[HER + 7],
        (const float*)d_in[HER + 8], (const float*)d_in[HER + 9],
        (const float*)d_in[HER + 10], (const float*)d_in[HER + 11],
        (float*)d_out + NNODES * 3);
}
